// DropoutDPP_30056181137674
// MI455X (gfx1250) — matrix operand order repaired and checked by reading, not compiled, not run on hardware
//
#include <hip/hip_runtime.h>
#include <stdint.h>

// ---------------------------------------------------------------------------
// DPP-dropout pipeline for MI455X (gfx1250, wave32, WMMA + TDM).
//   x: (64, 512, 768) f32  ->  out = x * sum_mask[col] / n_samples
// Stages:
//   1) column sums                          (streaming reduce)
//   2) G = X^T X via bf16 hi/lo split WMMA, tiles staged by the Tensor Data
//      Mover (double-buffered async DMA into LDS, TENSORcnt-tracked)
//   3) corr matrix L + V = I                (elementwise)
//   4) Jacobi eigensolver (persistent, grid barrier)
//   5) DPP sampler: per-sample K = (V z)V^T on the WMMA path, then
//      sequential rank-1 conditioning
//   6) streaming scale (HBM-bound, float4 + prefetch)
// ---------------------------------------------------------------------------

typedef __attribute__((ext_vector_type(16))) __bf16 v16bf;
typedef __attribute__((ext_vector_type(8)))  float  v8f;
typedef __attribute__((ext_vector_type(4)))  unsigned int u32x4;
typedef __attribute__((ext_vector_type(8)))  int    i32x8;
typedef __attribute__((ext_vector_type(4)))  int    i32x4;

constexpr int P       = 768;
constexpr int NROWS   = 32768;          // 64*512
constexpr int NBLK    = 48;             // persistent-kernel grid (must be co-resident)
constexpr int PTHREADS= 256;
constexpr int SWEEPS  = 10;             // Jacobi sweeps
constexpr int MAX_N   = 100;
constexpr float MAX_FRAC = 0.4f;

// workspace layout (float units)
constexpr size_t OFF_CTRL = 0;                       // 64 ints/floats control block
constexpr size_t OFF_MU   = 64;                      // column sums S_c
constexpr size_t OFF_D    = OFF_MU + P;              // sqrt(diag(cov))
constexpr size_t OFF_W    = OFF_D + P;               // eigenvalues (clipped)
constexpr size_t OFF_Z    = OFF_W + P;               // eigenvector bernoulli mask
constexpr size_t OFF_SEL  = OFF_Z + P;               // per-sample selection
constexpr size_t OFF_SM   = OFF_SEL + P;             // accumulated mask
constexpr size_t OFF_ROWB = OFF_SM + P;              // staged row K[idx,:]
constexpr size_t OFF_RC   = OFF_ROWB + P;            // Jacobi cos (384)
constexpr size_t OFF_RS   = OFF_RC + P / 2;          // Jacobi sin (384)
constexpr size_t OFF_L    = OFF_RS + P / 2;          // G -> L -> Jacobi in-place (p*p)
constexpr size_t OFF_V    = OFF_L + (size_t)P * P;   // eigenvectors (p*p)
constexpr size_t OFF_K    = OFF_V + (size_t)P * P;   // projection kernel (p*p)

// ---- deterministic counter-based RNG (JAX threefry not replicated) --------
__device__ inline unsigned int mix64to32(unsigned long long h) {
  h ^= h >> 33; h *= 0xff51afd7ed558ccdull;
  h ^= h >> 33; h *= 0xc4ceb9fe1a85ec53ull;
  h ^= h >> 33;
  return (unsigned int)h;
}
__device__ inline float rng_uniform(unsigned int a, unsigned int b, unsigned int c) {
  unsigned long long h = ((unsigned long long)a << 40) ^ ((unsigned long long)b << 20)
                       ^ (unsigned long long)c ^ 0x9E3779B97F4A7C15ull;
  h *= 0x9E3779B97F4A7C15ull;
  return (mix64to32(h) >> 8) * (1.0f / 16777216.0f);   // [0,1)
}

// ---- grid-wide spin barrier for persistent kernels ------------------------
__device__ inline void grid_barrier(int* cnt, volatile int* gen) {
  __threadfence();
  __syncthreads();
  if (threadIdx.x == 0) {
    int g = *gen;
    if (atomicAdd(cnt, 1) == (int)gridDim.x - 1) {
      atomicExch(cnt, 0);
      __threadfence();
      atomicAdd((int*)gen, 1);
    } else {
      while (*gen == g) { __builtin_amdgcn_s_sleep(1); }
    }
    __threadfence();
  }
  __syncthreads();
}

// ---- Tensor Data Mover: async 2-D f32 tile -> LDS (D# per ISA ch.8) -------
// tensor_dim0/1 set to the exact tile extent (tile is interior => no OOB),
// tensor_dim0_stride = row stride of the source matrix.
// This toolchain exposes the 6-arg builtin:
//   (u32x4 g0, i32x8 g1, i32x4 g2, i32x4 g3, i32x8, i32 cpol)
__device__ __forceinline__ void tdm_load_2d_f32(unsigned int lds_byte_off,
                                                const float* gptr,
                                                unsigned int tile_cols,
                                                unsigned int tile_rows,
                                                unsigned int row_stride_elems) {
  unsigned long long ga = (unsigned long long)(uintptr_t)gptr;
  u32x4 g0;
  g0[0] = 1u;                                       // count=1, user mode, no gather
  g0[1] = lds_byte_off;                             // LDS byte address
  g0[2] = (unsigned int)(ga & 0xffffffffu);         // global_addr[31:0]
  g0[3] = (unsigned int)((ga >> 32) & 0x1ffffffu)   // global_addr[56:32]
        | (2u << 30);                               // type=2 ("image")
  i32x8 g1;
  g1[0] = (int)(2u << 16);                          // wg_mask=0, data_size=2 (4B)
  g1[1] = (int)((tile_cols & 0xffffu) << 16);       // tensor_dim0[15:0] @ [63:48]
  g1[2] = (int)(((tile_cols >> 16) & 0xffffu)       // tensor_dim0[31:16]
        | ((tile_rows & 0xffffu) << 16));           // tensor_dim1[15:0]
  g1[3] = (int)(((tile_rows >> 16) & 0xffffu)       // tensor_dim1[31:16]
        | ((tile_cols & 0xffffu) << 16));           // tile_dim0
  g1[4] = (int)(tile_rows & 0xffffu);               // tile_dim1 ; tile_dim2=0
  g1[5] = (int)row_stride_elems;                    // tensor_dim0_stride[31:0]
  g1[6] = 0;                                        // stride hi / dim1_stride (2-D)
  g1[7] = 0;
  i32x4 gz4 = {0, 0, 0, 0};                         // groups 2/3 unused (<=2D)
  i32x8 gz8 = {0, 0, 0, 0, 0, 0, 0, 0};
  __builtin_amdgcn_tensor_load_to_lds(g0, g1, gz4, gz4, gz8, 0);
}

// ===========================================================================
// 1) column sums: mu[c] = sum_r X[r,c]
// ===========================================================================
__global__ __launch_bounds__(P)
void col_sum_kernel(const float* __restrict__ X, float* __restrict__ mu) {
  const int c = threadIdx.x;
  float s = 0.f;
  for (int r = blockIdx.x; r < NROWS; r += gridDim.x)
    s += X[(size_t)r * P + c];
  atomicAdd(&mu[c], s);
}

// ===========================================================================
// 2) G = X^T X  via  bf16 hi/lo split WMMA (f32 accumulate).
//    Block tile 64x64, 8 waves (4 M-strips x 2 N-strips of 32), K-chunk 32.
//    Source tiles staged by double-buffered TDM DMA into LDS.
// ===========================================================================
__global__ __launch_bounds__(256)
void gram_wmma_kernel(const float* __restrict__ X, float* __restrict__ G) {
  __shared__ float  XaF[2][32][64];    // TDM landing buffers (A operand)
  __shared__ float  XbF[2][32][64];    // TDM landing buffers (B operand)
  __shared__ __bf16 Ahi[64][34];
  __shared__ __bf16 Alo[64][34];
  __shared__ __bf16 Bhi[64][34];
  __shared__ __bf16 Blo[64][34];

  const int i0   = blockIdx.y * 64;
  const int j0   = blockIdx.x * 64;
  const int tid  = threadIdx.x;
  const int lane = tid & 31;
  const int wid  = tid >> 5;      // 0..7
  const int wm   = wid & 3;       // 16-row strip within tile
  const int wn   = wid >> 2;      // 0..1 : 32-col strip

  const unsigned int ldsA0 = (unsigned int)(uintptr_t)&XaF[0][0][0];
  const unsigned int ldsA1 = (unsigned int)(uintptr_t)&XaF[1][0][0];
  const unsigned int ldsB0 = (unsigned int)(uintptr_t)&XbF[0][0][0];
  const unsigned int ldsB1 = (unsigned int)(uintptr_t)&XbF[1][0][0];

  v8f acc0 = {};
  v8f acc1 = {};

  constexpr int NK = NROWS / 32;       // 1024 K-chunks

  // prologue: DMA first tile pair into buffer 0 (wave 0 issues; EXEC-independent
  // tensor ops must be skipped by wave-uniform branch, not by mask)
  if (wid == 0) {
    tdm_load_2d_f32(ldsA0, X + (size_t)0 * P + i0, 64u, 32u, (unsigned)P);
    tdm_load_2d_f32(ldsB0, X + (size_t)0 * P + j0, 64u, 32u, (unsigned)P);
  }

  for (int kb = 0; kb < NK; ++kb) {
    const int buf = kb & 1;
    if (wid == 0) {
      if (kb + 1 < NK) {
        const size_t krow = (size_t)(kb + 1) * 32;
        const unsigned int la = (kb & 1) ? ldsA0 : ldsA1;   // next buffer
        const unsigned int lb = (kb & 1) ? ldsB0 : ldsB1;
        tdm_load_2d_f32(la, X + krow * P + i0, 64u, 32u, (unsigned)P);
        tdm_load_2d_f32(lb, X + krow * P + j0, 64u, 32u, (unsigned)P);
        // 2 newly issued outstanding; <=2 means current tile pair landed
        __builtin_amdgcn_s_wait_tensorcnt(2);
      } else {
        __builtin_amdgcn_s_wait_tensorcnt(0);
      }
    }
    __syncthreads();   // current f32 tiles visible; prev fragments consumed

    // convert f32 -> bf16 hi+lo, transposed into [feature][k] layout
    #pragma unroll
    for (int it = 0; it < 8; ++it) {
      int e  = tid + it * 256;    // 0..2047
      int kk = e >> 6;            // 0..31
      int m  = e & 63;            // 0..63
      float xa = XaF[buf][kk][m];
      __bf16 ah = (__bf16)xa;
      Ahi[m][kk] = ah;
      Alo[m][kk] = (__bf16)(xa - (float)ah);
      float xb = XbF[buf][kk][m];
      __bf16 bh = (__bf16)xb;
      Bhi[m][kk] = bh;
      Blo[m][kk] = (__bf16)(xb - (float)bh);
    }
    __syncthreads();

    // A fragment (16x32 bf16): lane<16 -> K 0..7,16..23 ; lane>=16 -> +8
    // B fragment (32x16 bf16): lane<16 -> K 0..7,16..23 ; lane>=16 -> +8
    v16bf a_hi, a_lo, b0_hi, b0_lo, b1_hi, b1_lo;
    const int am   = wm * 16 + (lane & 15);
    const int bn0  = wn * 32 + (lane & 15);
    const int bn1  = bn0 + 16;
    const int khiA = (lane >= 16) ? 8 : 0;
    const int khiB = (lane >= 16) ? 16 : 0;
    #pragma unroll
    for (int e = 0; e < 16; ++e) {
      int ka = ((e < 8) ? e : (e + 8)) + khiA;
      a_hi[e]  = Ahi[am][ka];
      a_lo[e]  = Alo[am][ka];
      int kb2 = ka;
      b0_hi[e] = Bhi[bn0][kb2];
      b0_lo[e] = Blo[bn0][kb2];
      b1_hi[e] = Bhi[bn1][kb2];
      b1_lo[e] = Blo[bn1][kb2];
    }

    // 3-term bf16 emulation of f32 product: hi*hi + hi*lo + lo*hi
    acc0 = __builtin_amdgcn_wmma_f32_16x16x32_bf16(false, a_hi, false, b0_hi, (short)0, acc0, false, false);
    acc0 = __builtin_amdgcn_wmma_f32_16x16x32_bf16(false, a_hi, false, b0_lo, (short)0, acc0, false, false);
    acc0 = __builtin_amdgcn_wmma_f32_16x16x32_bf16(false, a_lo, false, b0_hi, (short)0, acc0, false, false);
    acc1 = __builtin_amdgcn_wmma_f32_16x16x32_bf16(false, a_hi, false, b1_hi, (short)0, acc1, false, false);
    acc1 = __builtin_amdgcn_wmma_f32_16x16x32_bf16(false, a_hi, false, b1_lo, (short)0, acc1, false, false);
    acc1 = __builtin_amdgcn_wmma_f32_16x16x32_bf16(false, a_lo, false, b1_hi, (short)0, acc1, false, false);
  }

  // C/D layout: VGPR v -> row v (lanes 0-15) / v+8 (lanes 16-31); col = lane%16
  const int col0  = j0 + wn * 32 + (lane & 15);
  const int rbase = i0 + wm * 16 + ((lane >= 16) ? 8 : 0);
  #pragma unroll
  for (int v = 0; v < 8; ++v) {
    G[(size_t)(rbase + v) * P + col0]        = acc0[v];
    G[(size_t)(rbase + v) * P + (col0 + 16)] = acc1[v];
  }
}

// ===========================================================================
// 3) d[i] = sqrt(clip(cov_ii, 1e-12));   L = cov / (d d^T) in place; V = I
// ===========================================================================
__global__ __launch_bounds__(256)
void dvec_kernel(float* ws_f) {
  int i = blockIdx.x * blockDim.x + threadIdx.x;
  if (i >= P) return;
  const float* G  = ws_f + OFF_L;
  const float* mu = ws_f + OFF_MU;
  float cov = G[(size_t)i * P + i] - mu[i] * mu[i] * (1.0f / (float)NROWS);
  ws_f[OFF_D + i] = sqrtf(fmaxf(cov, 1e-12f));
}

__global__ __launch_bounds__(256)
void corr_init_kernel(float* ws_f) {
  size_t e = (size_t)blockIdx.x * blockDim.x + threadIdx.x;
  if (e >= (size_t)P * P) return;
  int i = (int)(e / P), j = (int)(e % P);
  float* L  = ws_f + OFF_L;
  float* Vm = ws_f + OFF_V;
  const float* mu = ws_f + OFF_MU;
  const float* d  = ws_f + OFF_D;
  float cov = L[e] - mu[i] * mu[j] * (1.0f / (float)NROWS);
  L[e]  = cov / (d[i] * d[j]);
  Vm[e] = (i == j) ? 1.0f : 0.0f;
}

// ===========================================================================
// 4) cyclic two-sided Jacobi (round-robin ordering), persistent kernel.
//    A (=L buffer) diagonalized in place; V accumulates eigenvectors (cols).
// ===========================================================================
__device__ inline int slot_player(int j, int r) {
  return (j == 0) ? 0 : 1 + ((j - 1 + r) % (P - 1));
}

__global__ __launch_bounds__(PTHREADS)
void jacobi_kernel(float* ws_f) {
  float* A  = ws_f + OFF_L;
  float* Vm = ws_f + OFF_V;
  float* rc = ws_f + OFF_RC;
  float* rs = ws_f + OFF_RS;
  int*   cnt = (int*)ws_f + 0;
  volatile int* gen = (volatile int*)((int*)ws_f + 1);

  const int gsz = gridDim.x * blockDim.x;
  const int gid = blockIdx.x * blockDim.x + threadIdx.x;
  const int half = P / 2;

  for (int sweep = 0; sweep < SWEEPS; ++sweep) {
    for (int r = 0; r < P - 1; ++r) {
      // phase 1: rotation angles for 384 disjoint pairs
      for (int t = gid; t < half; t += gsz) {
        int a = slot_player(t, r);
        int b = slot_player(P - 1 - t, r);
        int p_ = a < b ? a : b;
        int q_ = a < b ? b : a;
        float app = A[(size_t)p_ * P + p_];
        float aqq = A[(size_t)q_ * P + q_];
        float apq = A[(size_t)p_ * P + q_];
        float c = 1.0f, s = 0.0f;
        if (fabsf(apq) > 1e-20f) {
          float tau = (aqq - app) / (2.0f * apq);
          float tt  = ((tau >= 0.0f) ? 1.0f : -1.0f) / (fabsf(tau) + sqrtf(1.0f + tau * tau));
          c = rsqrtf(1.0f + tt * tt);
          s = tt * c;
        }
        rc[t] = c; rs[t] = s;
      }
      grid_barrier(cnt, gen);
      // phase 2: row rotations (rows p,q disjoint across pairs)
      for (int e = gid; e < half * P; e += gsz) {
        int t = e / P, j = e % P;
        int a = slot_player(t, r);
        int b = slot_player(P - 1 - t, r);
        int p_ = a < b ? a : b;
        int q_ = a < b ? b : a;
        float c = rc[t], s = rs[t];
        float apj = A[(size_t)p_ * P + j];
        float aqj = A[(size_t)q_ * P + j];
        A[(size_t)p_ * P + j] = c * apj - s * aqj;
        A[(size_t)q_ * P + j] = s * apj + c * aqj;
      }
      grid_barrier(cnt, gen);
      // phase 3: column rotations + eigenvector update
      for (int e = gid; e < half * P; e += gsz) {
        int t = e / P, i = e % P;
        int a = slot_player(t, r);
        int b = slot_player(P - 1 - t, r);
        int p_ = a < b ? a : b;
        int q_ = a < b ? b : a;
        float c = rc[t], s = rs[t];
        float aip = A[(size_t)i * P + p_];
        float aiq = A[(size_t)i * P + q_];
        A[(size_t)i * P + p_] = c * aip - s * aiq;
        A[(size_t)i * P + q_] = s * aip + c * aiq;
        float vip = Vm[(size_t)i * P + p_];
        float viq = Vm[(size_t)i * P + q_];
        Vm[(size_t)i * P + p_] = c * vip - s * viq;
        Vm[(size_t)i * P + q_] = s * vip + c * viq;
      }
      grid_barrier(cnt, gen);
    }
  }
  // eigenvalues, clipped at 0 (reference: jnp.clip(w, 0.0))
  for (int e = gid; e < P; e += gsz) {
    float d = A[(size_t)e * P + e];
    ws_f[OFF_W + e] = d > 0.0f ? d : 0.0f;
  }
}

// ===========================================================================
// 5) DPP sampler: fully on-device data-dependent loop (max 100 samples).
//    Projection kernel K = (V diag(z)) V^T built with bf16 hi/lo WMMA tiles
//    (144 64x64 tiles = 3 per persistent block), then sequential rank-1
//    conditioning with grid barriers.
// ===========================================================================
__global__ __launch_bounds__(PTHREADS)
void dpp_sampler_kernel(float* ws_f) {
  __shared__ __bf16 Ahi[64][34];
  __shared__ __bf16 Alo[64][34];
  __shared__ __bf16 Bhi[64][34];
  __shared__ __bf16 Blo[64][34];

  int*   cnt = (int*)ws_f + 0;
  volatile int*   gen = (volatile int*)((int*)ws_f + 1);
  volatile int*   vci = (volatile int*)ws_f;
  volatile float* vcf = (volatile float*)ws_f;

  const float* Vm = ws_f + OFF_V;
  const float* w  = ws_f + OFF_W;
  float* z    = ws_f + OFF_Z;
  float* sel  = ws_f + OFF_SEL;
  float* sm   = ws_f + OFF_SM;
  float* rowb = ws_f + OFF_ROWB;
  float* K    = ws_f + OFF_K;

  const int gsz  = gridDim.x * blockDim.x;
  const int gid  = blockIdx.x * blockDim.x + threadIdx.x;
  const int tid  = threadIdx.x;
  const int lane = tid & 31;
  const int wid  = tid >> 5;
  const int wm   = wid & 3;
  const int wn   = wid >> 2;
  constexpr int NTILE = (P / 64) * (P / 64);   // 144

  for (int s = 0; s < MAX_N; ++s) {
    if (s > 0 && vcf[9] >= MAX_FRAC) break;     // uniform across grid

    // phase 1: eigenvector bernoullis, clear selection
    for (int e = gid; e < P; e += gsz) {
      float lam = w[e];
      float u = rng_uniform(42u, (unsigned)s, (unsigned)e);
      z[e]   = (u < lam / (1.0f + lam)) ? 1.0f : 0.0f;
      sel[e] = 0.0f;
    }
    grid_barrier(cnt, gen);

    // K = (V * diag(z)) V^T  -- WMMA-tiled (z binary => hi/lo split exact)
    for (int tile = blockIdx.x; tile < NTILE; tile += gridDim.x) {
      const int i0 = (tile / (P / 64)) * 64;
      const int j0 = (tile % (P / 64)) * 64;
      v8f acc0 = {};
      v8f acc1 = {};
      for (int e0 = 0; e0 < P; e0 += 32) {
        __syncthreads();
        // stage: A[m][kk] = V[i0+m, e0+kk] * z[e0+kk];  B[m][kk] = V[j0+m, e0+kk]
        #pragma unroll
        for (int it = 0; it < 8; ++it) {
          int e  = tid + it * 256;    // 0..2047
          int m  = e >> 5;            // 0..63 (kk contiguous per thread group)
          int kk = e & 31;            // 0..31
          float zv = z[e0 + kk];
          float va = Vm[(size_t)(i0 + m) * P + (e0 + kk)] * zv;
          __bf16 ah = (__bf16)va;
          Ahi[m][kk] = ah;
          Alo[m][kk] = (__bf16)(va - (float)ah);
          float vb = Vm[(size_t)(j0 + m) * P + (e0 + kk)];
          __bf16 bh = (__bf16)vb;
          Bhi[m][kk] = bh;
          Blo[m][kk] = (__bf16)(vb - (float)bh);
        }
        __syncthreads();

        v16bf a_hi, a_lo, b0_hi, b0_lo, b1_hi, b1_lo;
        const int am   = wm * 16 + (lane & 15);
        const int bn0  = wn * 32 + (lane & 15);
        const int bn1  = bn0 + 16;
        const int khiA = (lane >= 16) ? 8 : 0;
        const int khiB = (lane >= 16) ? 16 : 0;
        #pragma unroll
        for (int e = 0; e < 16; ++e) {
          int ka = ((e < 8) ? e : (e + 8)) + khiA;
          a_hi[e]  = Ahi[am][ka];
          a_lo[e]  = Alo[am][ka];
          int kb = ka;
          b0_hi[e] = Bhi[bn0][kb];
          b0_lo[e] = Blo[bn0][kb];
          b1_hi[e] = Bhi[bn1][kb];
          b1_lo[e] = Blo[bn1][kb];
        }
        acc0 = __builtin_amdgcn_wmma_f32_16x16x32_bf16(false, a_hi, false, b0_hi, (short)0, acc0, false, false);
        acc0 = __builtin_amdgcn_wmma_f32_16x16x32_bf16(false, a_hi, false, b0_lo, (short)0, acc0, false, false);
        acc0 = __builtin_amdgcn_wmma_f32_16x16x32_bf16(false, a_lo, false, b0_hi, (short)0, acc0, false, false);
        acc1 = __builtin_amdgcn_wmma_f32_16x16x32_bf16(false, a_hi, false, b1_hi, (short)0, acc1, false, false);
        acc1 = __builtin_amdgcn_wmma_f32_16x16x32_bf16(false, a_hi, false, b1_lo, (short)0, acc1, false, false);
        acc1 = __builtin_amdgcn_wmma_f32_16x16x32_bf16(false, a_lo, false, b1_hi, (short)0, acc1, false, false);
      }
      const int col0  = j0 + wn * 32 + (lane & 15);
      const int rbase = i0 + wm * 16 + ((lane >= 16) ? 8 : 0);
      #pragma unroll
      for (int v = 0; v < 8; ++v) {
        K[(size_t)(rbase + v) * P + col0]        = acc0[v];
        K[(size_t)(rbase + v) * P + (col0 + 16)] = acc1[v];
      }
    }
    if (gid == 0) {
      int kc = 0;
      for (int e = 0; e < P; ++e) kc += (z[e] != 0.0f);
      vci[4] = kc;
    }
    grid_barrier(cnt, gen);
    const int kc = vci[4];

    // phase 2: sequential projection-DPP conditioning
    for (int step = 0; step < P; ++step) {
      if (step >= kc) break;                    // uniform
      if (gid == 0) {
        float ssum = 0.0f;
        for (int d = 0; d < P; ++d) {
          float dv = K[(size_t)d * P + d];
          ssum += (dv > 0.0f ? dv : 0.0f);
        }
        float u = rng_uniform(1337u, (unsigned)(s * 1024 + step), 7u) * fmaxf(ssum, 1e-12f);
        float csum = 0.0f; int idx = P - 1;
        for (int d = 0; d < P; ++d) {
          float dv = K[(size_t)d * P + d];
          csum += (dv > 0.0f ? dv : 0.0f);
          if (u < csum) { idx = d; break; }
        }
        vci[3] = idx;
        vcf[8] = fmaxf(K[(size_t)idx * P + idx], 1e-12f);
        sel[idx] = 1.0f;
      }
      grid_barrier(cnt, gen);
      const int   idx = vci[3];
      const float inv = 1.0f / vcf[8];
      for (int e = gid; e < P; e += gsz) rowb[e] = K[(size_t)idx * P + e];
      grid_barrier(cnt, gen);
      // K stays symmetric: K[:,idx] == rowb
      for (int e = gid; e < P * P; e += gsz) {
        int i = e / P, j = e % P;
        K[e] -= rowb[i] * rowb[j] * inv;
      }
      grid_barrier(cnt, gen);
    }

    // accumulate mask, update coverage fraction + sample count
    for (int e = gid; e < P; e += gsz) sm[e] += sel[e];
    grid_barrier(cnt, gen);
    if (gid == 0) {
      int nz = 0;
      for (int e = 0; e < P; ++e) nz += (sm[e] != 0.0f);
      vcf[9] = (float)nz / (float)P;
      vci[2] = s + 1;
    }
    grid_barrier(cnt, gen);
  }
}

// ===========================================================================
// 6) out = x * sum_mask[col] / n_samples  (HBM-bound, float4 + prefetch)
// ===========================================================================
__global__ __launch_bounds__(256)
void scale_kernel(const float* __restrict__ X, const float* __restrict__ sm,
                  const int* __restrict__ ci, float* __restrict__ out) {
  const size_t total4 = (size_t)NROWS * P / 4;
  size_t g = (size_t)blockIdx.x * blockDim.x + threadIdx.x;
  if (g >= total4) return;
  const float4* X4 = (const float4*)X;
  __builtin_prefetch(&X4[g + 4096], 0, 1);
  int n = ci[2]; if (n < 1) n = 1;
  float inv = 1.0f / (float)n;
  float4 xv = X4[g];
  int c = (int)((g % (P / 4)) * 4);
  float4 o;
  o.x = xv.x * sm[c + 0] * inv;
  o.y = xv.y * sm[c + 1] * inv;
  o.z = xv.z * sm[c + 2] * inv;
  o.w = xv.w * sm[c + 3] * inv;
  ((float4*)out)[g] = o;
}

// ===========================================================================
extern "C" void kernel_launch(void* const* d_in, const int* in_sizes, int n_in,
                              void* d_out, int out_size, void* d_ws, size_t ws_size,
                              hipStream_t stream) {
  (void)in_sizes; (void)n_in; (void)out_size; (void)ws_size;
  const float* X   = (const float*)d_in[0];
  float*       out = (float*)d_out;
  float*       wsf = (float*)d_ws;

  // zero: control block + column sums (contiguous), and the accumulated mask
  (void)hipMemsetAsync(d_ws, 0, (OFF_MU + P) * sizeof(float), stream);
  (void)hipMemsetAsync(wsf + OFF_SM, 0, P * sizeof(float), stream);

  col_sum_kernel<<<64, P, 0, stream>>>(X, wsf + OFF_MU);

  dim3 ggrid(P / 64, P / 64);
  gram_wmma_kernel<<<ggrid, 256, 0, stream>>>(X, wsf + OFF_L);

  dvec_kernel<<<(P + 255) / 256, 256, 0, stream>>>(wsf);
  corr_init_kernel<<<(int)(((size_t)P * P + 255) / 256), 256, 0, stream>>>(wsf);

  jacobi_kernel<<<NBLK, PTHREADS, 0, stream>>>(wsf);
  dpp_sampler_kernel<<<NBLK, PTHREADS, 0, stream>>>(wsf);

  const size_t total4 = (size_t)NROWS * P / 4;
  scale_kernel<<<(int)((total4 + 255) / 256), 256, 0, stream>>>(
      X, wsf + OFF_SM, (const int*)wsf + 2, out);
}
